// BiCrossAttention_84018150244369
// MI455X (gfx1250) — matrix operand order repaired and checked by reading, not compiled, not run on hardware
//
#include <hip/hip_runtime.h>
#include <hip/hip_bf16.h>

#define NTOK   4096
#define DMODEL 512
#define NHEAD  8
#define DHEAD  64
#define EPSLN  1e-5f
#define LOG2E  1.4426950408889634f

typedef __attribute__((ext_vector_type(16))) __bf16 v16bf;
typedef __attribute__((ext_vector_type(8)))  __bf16 v8bf;
typedef __attribute__((ext_vector_type(8)))  float  v8f;
typedef int v4i __attribute__((vector_size(16)));

#ifndef __has_builtin
#define __has_builtin(x) 0
#endif
#if __has_builtin(__builtin_amdgcn_global_load_async_to_lds_b128) && \
    __has_builtin(__builtin_amdgcn_s_wait_asynccnt)
#define USE_ASYNC_LDS 1
#else
#define USE_ASYNC_LDS 0
#endif

union AB16 { v16bf v; v8bf h[2]; };

// A operand (16x32 bf16): lane m(0..15)/m+16 split; two contiguous 16B loads.
__device__ __forceinline__ v16bf load_matA(const __bf16* __restrict__ base, int ld,
                                           int m0, int k0, int lane) {
  int m = lane & 15, hh = lane >> 4;
  const __bf16* p = base + (size_t)(m0 + m) * ld + k0 + hh * 16;
  AB16 u;
  u.h[0] = *(const v8bf*)p;
  u.h[1] = *(const v8bf*)(p + 8);
  return u.v;
}

// B operand (32x16 bf16): lane n holds column n, 16 contiguous K values (32B load).
__device__ __forceinline__ v16bf load_matB(const __bf16* __restrict__ base, int ld,
                                           int n0, int k0, int lane) {
  int n = lane & 15, hh = lane >> 4;
  return *(const v16bf*)(base + (size_t)(n0 + n) * ld + k0 + hh * 16);
}

__device__ __forceinline__ v8f wmma_bf16(v16bf a, v16bf b, v8f c) {
  return __builtin_amdgcn_wmma_f32_16x16x32_bf16(false, a, false, b, (short)0, c,
                                                 false, false);
}

// 16-byte global -> LDS staging; prefers CDNA5 async-to-LDS DMA (ASYNCcnt).
__device__ __forceinline__ void stage16(const __bf16* __restrict__ g, __bf16* l) {
#if USE_ASYNC_LDS
  __builtin_amdgcn_global_load_async_to_lds_b128(
      (__attribute__((address_space(1))) v4i*)g,
      (__attribute__((address_space(3))) v4i*)l, 0, 0);
#else
  *(v8bf*)l = *(const v8bf*)g;
#endif
}

__device__ __forceinline__ void wait_async_stage() {
#if USE_ASYNC_LDS
  __builtin_amdgcn_s_wait_asynccnt(0);
#endif
}

// ---------------- conversion kernels ----------------
__global__ void __launch_bounds__(256)
f32_to_bf16_kernel(const float* __restrict__ src, __bf16* __restrict__ dst, int n) {
  int i = blockIdx.x * blockDim.x + threadIdx.x;
  if (i < n) dst[i] = (__bf16)src[i];
}

__global__ void __launch_bounds__(256)
transpose_bf16_kernel(const float* __restrict__ W, __bf16* __restrict__ WT,
                      int din, int dout) {
  int i = blockIdx.x * blockDim.x + threadIdx.x;
  if (i < din * dout) {
    int r = i / dout, c = i % dout;
    WT[(size_t)c * din + r] = (__bf16)W[i];
  }
}

// ---------------- projection GEMM: out = (X @ W + b) * scale  ----------------
__global__ void __launch_bounds__(128)
proj_gemm_kernel(const __bf16* __restrict__ X, const __bf16* __restrict__ WT,
                 const float* __restrict__ bias, __bf16* __restrict__ Out,
                 int din, float scale, int transpose_out) {
  const int lane = threadIdx.x & 31;
  const int wg   = blockIdx.x * 4 + (threadIdx.x >> 5);
  const int m0 = (wg >> 5) * 16;
  const int n0 = (wg & 31) * 16;
  const int n = lane & 15, hh = lane >> 4;

  v8f acc = {};
  for (int k0 = 0; k0 < din; k0 += 32) {
    v16bf a = load_matA(X, din, m0, k0, lane);
    v16bf b = load_matB(WT, din, n0, k0, lane);
    acc = wmma_bf16(a, b, acc);
  }
  float bv = bias[n0 + n];
  if (transpose_out) {
#pragma unroll
    for (int r = 0; r < 8; ++r)
      Out[(size_t)(n0 + n) * NTOK + m0 + hh * 8 + r] = (__bf16)((acc[r] + bv) * scale);
  } else {
#pragma unroll
    for (int r = 0; r < 8; ++r)
      Out[(size_t)(m0 + hh * 8 + r) * DMODEL + n0 + n] = (__bf16)((acc[r] + bv) * scale);
  }
}

// ---------------- flash attention -------------------------------------------
// 256-thread block = 8 waves = 8 Q-tiles (128 query rows) of ONE head.
// K/V chunks (32 keys) are cooperatively staged into double-buffered LDS, so
// each block reads K/V once (8x less L2 traffic than per-wave streaming).
// Q is pre-scaled by (1/8)*log2(e): softmax runs in the base-2 domain.
__global__ void __launch_bounds__(256)
attn_kernel(const __bf16* __restrict__ Q, const __bf16* __restrict__ K,
            const __bf16* __restrict__ VT, __bf16* __restrict__ Out) {
  __shared__ __attribute__((aligned(16))) __bf16 kbuf[2][32 * 72]; // keys x d, pad 72
  __shared__ __attribute__((aligned(16))) __bf16 vbuf[2][64 * 40]; // d x keys, pad 40
  __shared__ __attribute__((aligned(16))) __bf16 pbuf[8][16][40];  // P bounce / wave

  const int lane = threadIdx.x & 31;
  const int wid  = threadIdx.x >> 5;
  const int m0   = (blockIdx.x * 8 + wid) * 16;
  const int hbase = blockIdx.y * DHEAD;
  const int n = lane & 15, hh = lane >> 4;

  // staging roles (per thread, uniform across loop)
  const int krow = threadIdx.x >> 3, kseg = threadIdx.x & 7;  // 32 rows x 8*16B
  const int vrow = threadIdx.x >> 2, vseg = threadIdx.x & 3;  // 64 rows x 4*16B

  // Q tile A-operands (DH = 64 -> two 32-slices), loaded once
  v16bf aq0 = load_matA(Q, DMODEL, m0, hbase, lane);
  v16bf aq1 = load_matA(Q, DMODEL, m0, hbase + 32, lane);

  float mrun[8], lrun[8];
  v8f o0 = {}, o1 = {}, o2 = {}, o3 = {};
#pragma unroll
  for (int r = 0; r < 8; ++r) { mrun[r] = -3.0e38f; lrun[r] = 0.f; }

  // stage chunk 0 into buffer 0
  stage16(K + (size_t)krow * DMODEL + hbase + kseg * 8, &kbuf[0][krow * 72 + kseg * 8]);
  stage16(VT + (size_t)(hbase + vrow) * NTOK + vseg * 8, &vbuf[0][vrow * 40 + vseg * 8]);

  int p = 0;
  for (int j = 0; j < NTOK; j += 32) {
    wait_async_stage();
    __syncthreads();  // chunk j visible in buffer p; prev compute done

    if (j + 32 < NTOK) {  // stage next chunk into the other buffer (overlapped)
      stage16(K + (size_t)(j + 32 + krow) * DMODEL + hbase + kseg * 8,
              &kbuf[p ^ 1][krow * 72 + kseg * 8]);
      stage16(VT + (size_t)(hbase + vrow) * NTOK + j + 32 + vseg * 8,
              &vbuf[p ^ 1][vrow * 40 + vseg * 8]);
    }

    const __bf16* kb = kbuf[p];
    const __bf16* vb = vbuf[p];

    // scores S = Q K^T : two 16x16 f32 tiles (keys j..j+15, j+16..j+31)
    v16bf bk00 = load_matB(kb, 72, 0,  0,  lane);
    v16bf bk01 = load_matB(kb, 72, 0,  32, lane);
    v16bf bk10 = load_matB(kb, 72, 16, 0,  lane);
    v16bf bk11 = load_matB(kb, 72, 16, 32, lane);
    v8f s0 = {}, s1 = {};
    s0 = wmma_bf16(aq0, bk00, s0);
    s0 = wmma_bf16(aq1, bk01, s0);
    s1 = wmma_bf16(aq0, bk10, s1);
    s1 = wmma_bf16(aq1, bk11, s1);

    // online softmax (base-2): each query row lives in one 16-lane half
    float corr[8];
#pragma unroll
    for (int r = 0; r < 8; ++r) {
      float v = fmaxf(s0[r], s1[r]);
      v = fmaxf(v, __shfl_xor(v, 1, 32));
      v = fmaxf(v, __shfl_xor(v, 2, 32));
      v = fmaxf(v, __shfl_xor(v, 4, 32));
      v = fmaxf(v, __shfl_xor(v, 8, 32));
      float mnew = fmaxf(mrun[r], v);
      corr[r] = __builtin_exp2f(mrun[r] - mnew);
      mrun[r] = mnew;
    }
#pragma unroll
    for (int r = 0; r < 8; ++r) {
      float e0 = __builtin_exp2f(s0[r] - mrun[r]);
      float e1 = __builtin_exp2f(s1[r] - mrun[r]);
      float ps = e0 + e1;
      ps += __shfl_xor(ps, 1, 32);
      ps += __shfl_xor(ps, 2, 32);
      ps += __shfl_xor(ps, 4, 32);
      ps += __shfl_xor(ps, 8, 32);
      lrun[r] = lrun[r] * corr[r] + ps;
      o0[r] *= corr[r]; o1[r] *= corr[r]; o2[r] *= corr[r]; o3[r] *= corr[r];
      pbuf[wid][hh * 8 + r][n]      = (__bf16)e0;   // P tile -> LDS, row-major
      pbuf[wid][hh * 8 + r][16 + n] = (__bf16)e1;
    }
    asm volatile("" ::: "memory");  // compile-time fence; DS ops are HW in-order

    // P (16x32) as A operand from LDS; V B-operands from staged LDS tile
    v16bf pa  = load_matA(&pbuf[wid][0][0], 40, 0, 0, lane);
    v16bf bv0 = load_matB(vb, 40, 0,  0, lane);
    v16bf bv1 = load_matB(vb, 40, 16, 0, lane);
    v16bf bv2 = load_matB(vb, 40, 32, 0, lane);
    v16bf bv3 = load_matB(vb, 40, 48, 0, lane);
    o0 = wmma_bf16(pa, bv0, o0);
    o1 = wmma_bf16(pa, bv1, o1);
    o2 = wmma_bf16(pa, bv2, o2);
    o3 = wmma_bf16(pa, bv3, o3);

    p ^= 1;
  }

#pragma unroll
  for (int r = 0; r < 8; ++r) {
    float inv = 1.0f / lrun[r];
    size_t row = (size_t)(m0 + hh * 8 + r) * DMODEL + hbase;
    Out[row +  0 + n] = (__bf16)(o0[r] * inv);
    Out[row + 16 + n] = (__bf16)(o1[r] * inv);
    Out[row + 32 + n] = (__bf16)(o2[r] * inv);
    Out[row + 48 + n] = (__bf16)(o3[r] * inv);
  }
}

// ---------------- fused output projections + residual sum + LayerNorm -------
__global__ void __launch_bounds__(512)
outproj_ln_kernel(const __bf16* __restrict__ o23, const __bf16* __restrict__ o32,
                  const __bf16* __restrict__ WoT23, const __bf16* __restrict__ WoT32,
                  const float* __restrict__ bo23, const float* __restrict__ bo32,
                  const float* __restrict__ gamma, const float* __restrict__ beta,
                  float* __restrict__ Out) {
  __shared__ float ybuf[16][DMODEL + 8];
  const int lane = threadIdx.x & 31;
  const int wid  = threadIdx.x >> 5;        // 16 waves
  const int m0   = blockIdx.x * 16;
  const int col0 = wid * 32;
  const int n = lane & 15, hh = lane >> 4;

  v8f acc0 = {}, acc1 = {};
  for (int k0 = 0; k0 < DMODEL; k0 += 32) {
    v16bf a  = load_matA(o23, DMODEL, m0, k0, lane);
    v16bf b0 = load_matB(WoT23, DMODEL, col0,      k0, lane);
    v16bf b1 = load_matB(WoT23, DMODEL, col0 + 16, k0, lane);
    acc0 = wmma_bf16(a, b0, acc0);
    acc1 = wmma_bf16(a, b1, acc1);
  }
  for (int k0 = 0; k0 < DMODEL; k0 += 32) {
    v16bf a  = load_matA(o32, DMODEL, m0, k0, lane);
    v16bf b0 = load_matB(WoT32, DMODEL, col0,      k0, lane);
    v16bf b1 = load_matB(WoT32, DMODEL, col0 + 16, k0, lane);
    acc0 = wmma_bf16(a, b0, acc0);
    acc1 = wmma_bf16(a, b1, acc1);
  }
  float bb0 = bo23[col0 + n]      + bo32[col0 + n];
  float bb1 = bo23[col0 + 16 + n] + bo32[col0 + 16 + n];
#pragma unroll
  for (int r = 0; r < 8; ++r) {
    ybuf[hh * 8 + r][col0 + n]      = acc0[r] + bb0;
    ybuf[hh * 8 + r][col0 + 16 + n] = acc1[r] + bb1;
  }
  __syncthreads();

  float s = 0.f, ss = 0.f;
#pragma unroll
  for (int i = 0; i < DMODEL / 32; ++i) {
    float x = ybuf[wid][lane + i * 32];
    s += x; ss += x * x;
  }
#pragma unroll
  for (int mask = 1; mask < 32; mask <<= 1) {
    s  += __shfl_xor(s,  mask, 32);
    ss += __shfl_xor(ss, mask, 32);
  }
  float mu   = s * (1.0f / DMODEL);
  float var  = ss * (1.0f / DMODEL) - mu * mu;
  float rstd = rsqrtf(var + EPSLN);
#pragma unroll
  for (int i = 0; i < DMODEL / 32; ++i) {
    int c = lane + i * 32;
    float x = ybuf[wid][c];
    Out[(size_t)(m0 + wid) * DMODEL + c] = (x - mu) * rstd * gamma[c] + beta[c];
  }
}

// ---------------- host launcher ----------------
extern "C" void kernel_launch(void* const* d_in, const int* in_sizes, int n_in,
                              void* d_out, int out_size, void* d_ws, size_t ws_size,
                              hipStream_t stream) {
  const float* h2d  = (const float*)d_in[0];
  const float* h3d  = (const float*)d_in[1];
  const float* Wq2d = (const float*)d_in[2];  const float* bq2d = (const float*)d_in[3];
  const float* Wk3d = (const float*)d_in[4];  const float* bk3d = (const float*)d_in[5];
  const float* Wv3d = (const float*)d_in[6];  const float* bv3d = (const float*)d_in[7];
  const float* Wq3d = (const float*)d_in[8];  const float* bq3d = (const float*)d_in[9];
  const float* Wk2d = (const float*)d_in[10]; const float* bk2d = (const float*)d_in[11];
  const float* Wv2d = (const float*)d_in[12]; const float* bv2d = (const float*)d_in[13];
  const float* Wo23 = (const float*)d_in[14]; const float* bo23 = (const float*)d_in[15];
  const float* Wo32 = (const float*)d_in[16]; const float* bo32 = (const float*)d_in[17];
  const float* gamma = (const float*)d_in[18]; const float* beta = (const float*)d_in[19];
  float* out = (float*)d_out;

  char* ws = (char*)d_ws;
  size_t off = 0;
  auto alloc = [&](size_t elems) {
    __bf16* p = (__bf16*)(ws + off);
    off += ((elems * 2 + 255) / 256) * 256;
    return p;
  };
  __bf16* h2d_bf = alloc((size_t)NTOK * 256);
  __bf16* h3d_bf = alloc((size_t)NTOK * 128);
  __bf16* WqT2d = alloc(512 * 256);
  __bf16* WkT3d = alloc(512 * 128);
  __bf16* WvT3d = alloc(512 * 128);
  __bf16* WqT3d = alloc(512 * 128);
  __bf16* WkT2d = alloc(512 * 256);
  __bf16* WvT2d = alloc(512 * 256);
  __bf16* WoT23 = alloc(512 * 512);
  __bf16* WoT32 = alloc(512 * 512);
  __bf16* Q2  = alloc((size_t)NTOK * DMODEL);
  __bf16* K3  = alloc((size_t)NTOK * DMODEL);
  __bf16* V3T = alloc((size_t)NTOK * DMODEL);
  __bf16* Q3  = alloc((size_t)NTOK * DMODEL);
  __bf16* K2  = alloc((size_t)NTOK * DMODEL);
  __bf16* V2T = alloc((size_t)NTOK * DMODEL);
  __bf16* o23 = alloc((size_t)NTOK * DMODEL);
  __bf16* o32 = alloc((size_t)NTOK * DMODEL);

  f32_to_bf16_kernel<<<(NTOK * 256) / 256, 256, 0, stream>>>(h2d, h2d_bf, NTOK * 256);
  f32_to_bf16_kernel<<<(NTOK * 128) / 256, 256, 0, stream>>>(h3d, h3d_bf, NTOK * 128);

  transpose_bf16_kernel<<<(256 * 512) / 256, 256, 0, stream>>>(Wq2d, WqT2d, 256, 512);
  transpose_bf16_kernel<<<(128 * 512) / 256, 256, 0, stream>>>(Wk3d, WkT3d, 128, 512);
  transpose_bf16_kernel<<<(128 * 512) / 256, 256, 0, stream>>>(Wv3d, WvT3d, 128, 512);
  transpose_bf16_kernel<<<(128 * 512) / 256, 256, 0, stream>>>(Wq3d, WqT3d, 128, 512);
  transpose_bf16_kernel<<<(256 * 512) / 256, 256, 0, stream>>>(Wk2d, WkT2d, 256, 512);
  transpose_bf16_kernel<<<(256 * 512) / 256, 256, 0, stream>>>(Wv2d, WvT2d, 256, 512);
  transpose_bf16_kernel<<<(512 * 512) / 256, 256, 0, stream>>>(Wo23, WoT23, 512, 512);
  transpose_bf16_kernel<<<(512 * 512) / 256, 256, 0, stream>>>(Wo32, WoT32, 512, 512);

  const float qs = 0.125f * LOG2E;  // 1/sqrt(DH) folded with log2(e) for exp2 softmax
  proj_gemm_kernel<<<2048, 128, 0, stream>>>(h2d_bf, WqT2d, bq2d, Q2,  256, qs,   0);
  proj_gemm_kernel<<<2048, 128, 0, stream>>>(h3d_bf, WkT3d, bk3d, K3,  128, 1.0f, 0);
  proj_gemm_kernel<<<2048, 128, 0, stream>>>(h3d_bf, WvT3d, bv3d, V3T, 128, 1.0f, 1);
  proj_gemm_kernel<<<2048, 128, 0, stream>>>(h3d_bf, WqT3d, bq3d, Q3,  128, qs,   0);
  proj_gemm_kernel<<<2048, 128, 0, stream>>>(h2d_bf, WkT2d, bk2d, K2,  256, 1.0f, 0);
  proj_gemm_kernel<<<2048, 128, 0, stream>>>(h2d_bf, WvT2d, bv2d, V2T, 256, 1.0f, 1);

  // 32 blocks x 8 Q-tiles = 256 Q-tiles per head; grid.y = head
  attn_kernel<<<dim3(32, NHEAD), 256, 0, stream>>>(Q2, K3, V3T, o23);
  attn_kernel<<<dim3(32, NHEAD), 256, 0, stream>>>(Q3, K2, V2T, o32);

  outproj_ln_kernel<<<NTOK / 16, 512, 0, stream>>>(o23, o32, WoT23, WoT32,
                                                   bo23, bo32, gamma, beta, out);
}
